// LocalSpatioTemporalAttention_58119497450438
// MI455X (gfx1250) — compile-verified, matrix operand order checked by reading, not run on hardware
//
#include <hip/hip_runtime.h>
#include <hip/hip_bf16.h>

typedef __attribute__((ext_vector_type(16))) __bf16       v16bf;
typedef __attribute__((ext_vector_type(8)))  float        v8f;
typedef __attribute__((ext_vector_type(4)))  unsigned int v4u;
typedef __attribute__((ext_vector_type(8)))  int          v8i;
typedef __attribute__((ext_vector_type(4)))  int          v4i;

#define T_   3
#define B_   2
#define Y_   192
#define X_   192
#define C_   128
#define H_   8
#define F_   16
#define HF_  128
#define S_   25
#define NROW (T_*B_*Y_*X_)   /* 221184 */
#define BYX  (B_*Y_*X_)      /* 73728  */
#define YX   (Y_*X_)         /* 36864  */

#if defined(__AMDGCN__) && __has_builtin(__builtin_amdgcn_tensor_load_to_lds) && \
    __has_builtin(__builtin_amdgcn_s_wait_tensorcnt)
#define USE_TDM 1
#else
#define USE_TDM 0
#endif

static __device__ __forceinline__ v8f wmma_bf16(v16bf a, v16bf b, v8f c) {
    return __builtin_amdgcn_wmma_f32_16x16x32_bf16(false, a, false, b, (short)0, c, false, false);
}

#if USE_TDM
// Tensor Data Mover: load an up-to-3D tile (8-byte elements) into LDS.
// D# bitfields per CDNA5 ISA §8 (group0: count/lds_addr/global_addr/type=2;
// group1: data_size=8B, tensor+tile dims, 48-bit strides; group2: tensor_dim2).
// Strides and dims are in 8-byte data_size units.
static __device__ __forceinline__ void tdm_load_to_lds(
    unsigned lds_addr, const void* gptr,
    unsigned td0, unsigned td1, unsigned td2,
    unsigned tile0, unsigned tile1, unsigned tile2,
    unsigned long long s0, unsigned long long s1)
{
    unsigned long long ga = (unsigned long long)(size_t)gptr;
    v4u g0;
    g0[0] = 1u;                                               // count=1 (valid user D#)
    g0[1] = lds_addr;                                         // LDS byte address
    g0[2] = (unsigned)ga;                                     // global_addr[31:0]
    g0[3] = ((unsigned)(ga >> 32) & 0x01FFFFFFu) | (2u << 30);// global_addr[56:32] | type=2
    v8i g1;
    g1[0] = (int)(3u << 16);                                  // data_size = 8B; wg_mask=0
    g1[1] = (int)((td0 & 0xFFFFu) << 16);                     // tensor_dim0[15:0]
    g1[2] = (int)((td0 >> 16) | ((td1 & 0xFFFFu) << 16));     // tensor_dim0[31:16]|dim1[15:0]
    g1[3] = (int)((td1 >> 16) | (tile0 << 16));               // tensor_dim1[31:16]|tile_dim0
    g1[4] = (int)(tile1 | (tile2 << 16));                     // tile_dim1 | tile_dim2
    g1[5] = (int)(unsigned)s0;                                // tensor_dim0_stride[31:0]
    g1[6] = (int)((unsigned)(s0 >> 32) |
                  ((unsigned)(s1 & 0xFFFFu) << 16));          // s0[47:32] | s1[15:0]
    g1[7] = (int)(unsigned)(s1 >> 16);                        // s1[47:16]
    v4i g2;
    g2[0] = (int)td2;                                         // tensor_dim2
    g2[1] = 0; g2[2] = 0; g2[3] = 0;
    v4i g3; g3[0] = 0; g3[1] = 0; g3[2] = 0; g3[3] = 0;
#if __has_include(<hip/amd_detail/amd_gfx1250_TDM.h>)
    v8i g4; g4[0]=0; g4[1]=0; g4[2]=0; g4[3]=0; g4[4]=0; g4[5]=0; g4[6]=0; g4[7]=0;
    __builtin_amdgcn_tensor_load_to_lds(g0, g1, g2, g3, g4, 0);   // clang-23 arity
#else
    __builtin_amdgcn_tensor_load_to_lds(g0, g1, g2, g3, 0);       // ROCm 7.2 arity
#endif
}
#endif  // USE_TDM

// ---------------------------------------------------------------------------
// Kernel 1: fused Q/K/V projection (bf16 WMMA, fp32 accumulate) + embeddings.
// ---------------------------------------------------------------------------
__global__ __launch_bounds__(256)
void qkv_proj_kernel(const float* __restrict__ frames,
                     const float* __restrict__ Wq,
                     const float* __restrict__ Wk,
                     const float* __restrict__ Wv,
                     const float* __restrict__ temp_emb,
                     const float* __restrict__ spatial_emb,
                     __bf16* __restrict__ Qb,
                     __bf16* __restrict__ Kb,
                     __bf16* __restrict__ Vb)
{
    __shared__ __attribute__((aligned(16))) __bf16 sA[64 * 128];
    __shared__ __attribute__((aligned(16))) __bf16 sW[128 * 128];
    __shared__ int sRowT[64];
    __shared__ int sRowS[64];

    const int tid     = threadIdx.x;
    const int rowBase = blockIdx.x * 64;

    for (int idx = tid; idx < 64 * 128; idx += 256)
        sA[idx] = (__bf16)frames[(size_t)rowBase * 128 + idx];

    if (tid < 64) {
        int row = rowBase + tid;
        int t   = row / BYX;
        int r3  = row % YX;
        int y = r3 / X_, x = r3 % X_;
        int yc = min(max(y, 2), Y_ - 3);
        int xc = min(max(x, 2), X_ - 3);
        sRowT[tid] = t;
        sRowS[tid] = (y - yc + 2) * 5 + (x - xc + 2);
    }

    if (blockIdx.x + 1 < gridDim.x)
        __builtin_prefetch(frames + ((size_t)rowBase + 64) * 128, 0, 0);

    const int wave = tid >> 5;
    const int lane = tid & 31;
    const int lr   = lane & 15;
    const int hi   = lane >> 4;

    for (int mat = 0; mat < 3; ++mat) {
        const float* W   = (mat == 0) ? Wq : ((mat == 1) ? Wk : Wv);
        __bf16*      dst = (mat == 0) ? Qb : ((mat == 1) ? Kb : Vb);

        __syncthreads();
        for (int idx = tid; idx < 128 * 128; idx += 256) {
            int n = idx >> 7, k = idx & 127;
            sW[idx] = (__bf16)W[k * 128 + n];
        }
        __syncthreads();

        for (int tile = wave; tile < 32; tile += 8) {
            int mt = tile >> 3, nt = tile & 7;
            v8f acc = {0.f, 0.f, 0.f, 0.f, 0.f, 0.f, 0.f, 0.f};
            #pragma unroll
            for (int kk = 0; kk < 4; ++kk) {
                const __bf16* ap = &sA[(mt * 16 + lr) * 128 + kk * 32 + hi * 8];
                const __bf16* bp = &sW[(nt * 16 + lr) * 128 + kk * 32 + hi * 8];
                v16bf a, b;
                #pragma unroll
                for (int i = 0; i < 8; ++i) {
                    a[i] = ap[i];  a[8 + i] = ap[16 + i];
                    b[i] = bp[i];  b[8 + i] = bp[16 + i];
                }
                acc = wmma_bf16(a, b, acc);
            }
            int col = nt * 16 + lr;
            #pragma unroll
            for (int r = 0; r < 8; ++r) {
                int   m = mt * 16 + r + hi * 8;
                float v = acc[r];
                if (mat < 2) {
                    v += temp_emb[sRowT[m] * HF_ + col];
                    if (mat == 0)
                        v += spatial_emb[sRowS[m] * HF_ + col];
                }
                dst[(size_t)(rowBase + m) * HF_ + col] = (__bf16)v;
            }
        }
    }
}

// ---------------------------------------------------------------------------
// Kernel 2: local spatio-temporal attention on WMMA, K/V patches via TDM.
// ---------------------------------------------------------------------------
__global__ __launch_bounds__(256)
void attn_kernel(const __bf16* Qb,
                 const __bf16* __restrict__ Kb,
                 const __bf16* __restrict__ Vb,
                 const float*  __restrict__ spatial_emb,
                 __bf16*       Ob)
{
    __shared__ __attribute__((aligned(16))) __bf16 sK[400 * 16];
    __shared__ __attribute__((aligned(16))) __bf16 sV[400 * 16];
    __shared__ __attribute__((aligned(16))) __bf16 sSE[32 * 16];
    __shared__ __attribute__((aligned(16))) float  sQs[8][16 * 32];
    __shared__ __attribute__((aligned(16))) __bf16 sP[8][16 * 32];

    const int tid  = threadIdx.x;
    const int wave = tid >> 5;
    const int lane = tid & 31;
    const int lr   = lane & 15;
    const int hi   = lane >> 4;

    const int bid = blockIdx.x;
    const int bb  = bid / 144;
    const int r2  = bid % 144;
    const int ty  = r2 / 12, tx = r2 % 12;
    const int y0  = ty * 16, x0 = tx * 16;
    const int py0 = min(max(y0 - 2, 0), Y_ - 20);
    const int px0 = min(max(x0 - 2, 0), X_ - 20);
    const float scale = 0.25f;

    for (int h = 0; h < H_; ++h) {
        __syncthreads();
        for (int idx = tid; idx < 32 * 16; idx += 256) {
            int s = idx >> 4, f = idx & 15;
            sSE[idx] = (s < S_) ? (__bf16)spatial_emb[s * HF_ + h * 16 + f] : (__bf16)0.f;
        }
        __syncthreads();

        for (int tq = 0; tq < T_; ++tq) {
            for (int ri = 0; ri < 2; ++ri) {
                const int yl = wave + ri * 8;
                const int y  = y0 + yl;
                const int yc = min(max(y, 2), Y_ - 3);

                // Q fragment (A layout): M = query x-index, K = F padded to 32.
                v16bf qa;
                {
                    const int qx = x0 + lr;
                    size_t qrow = ((size_t)(tq * B_ + bb) * Y_ + y) * X_ + qx;
                    const __bf16* qp = Qb + qrow * HF_ + h * 16 + hi * 8;
                    #pragma unroll
                    for (int i = 0; i < 8; ++i) { qa[i] = qp[i]; qa[8 + i] = (__bf16)0.f; }
                }

                // Qs = Q x semb^T (2 WMMAs), per-wave LDS.
                {
                    v8f c0 = {0.f,0.f,0.f,0.f,0.f,0.f,0.f,0.f};
                    v8f c1 = {0.f,0.f,0.f,0.f,0.f,0.f,0.f,0.f};
                    v16bf b0, b1;
                    #pragma unroll
                    for (int i = 0; i < 8; ++i) {
                        b0[i] = sSE[lr * 16 + hi * 8 + i];        b0[8 + i] = (__bf16)0.f;
                        b1[i] = sSE[(lr + 16) * 16 + hi * 8 + i]; b1[8 + i] = (__bf16)0.f;
                    }
                    c0 = wmma_bf16(qa, b0, c0);
                    c1 = wmma_bf16(qa, b1, c1);
                    #pragma unroll
                    for (int r = 0; r < 8; ++r) {
                        sQs[wave][(r + hi * 8) * 32 + lr]      = c0[r];
                        sQs[wave][(r + hi * 8) * 32 + lr + 16] = c1[r];
                    }
                }

                float m8[8], l8[8];
                v8f acc = {0.f,0.f,0.f,0.f,0.f,0.f,0.f,0.f};
                #pragma unroll
                for (int r = 0; r < 8; ++r) { m8[r] = -1e30f; l8[r] = 0.f; }

                for (int t = 0; t < T_; ++t) {
                    __syncthreads();   // previous K/V patch fully consumed
#if USE_TDM
                    // TDM: 3D tile (32B head slice) x (20 cols) x (20 rows),
                    // issued by wave 0 only; TENSORcnt drained before barrier.
                    if (wave == 0) {
                        const size_t base =
                            (((size_t)(t * B_ + bb) * Y_ + py0) * X_ + px0) * HF_ + h * 16;
                        tdm_load_to_lds((unsigned)(size_t)(void*)&sK[0], Kb + base,
                                        4, 20, 20,  4, 20, 20,
                                        (unsigned long long)(HF_ * 2 / 8),
                                        (unsigned long long)((size_t)X_ * HF_ * 2 / 8));
                        tdm_load_to_lds((unsigned)(size_t)(void*)&sV[0], Vb + base,
                                        4, 20, 20,  4, 20, 20,
                                        (unsigned long long)(HF_ * 2 / 8),
                                        (unsigned long long)((size_t)X_ * HF_ * 2 / 8));
                        __builtin_amdgcn_s_wait_tensorcnt(0);
                    }
#else
                    for (int pos = tid; pos < 400; pos += 256) {
                        int pr = pos / 20, pc = pos % 20;
                        size_t krow = ((size_t)(t * B_ + bb) * Y_ + (py0 + pr)) * X_ + (px0 + pc);
                        const __bf16* ks = Kb + krow * HF_ + h * 16;
                        const __bf16* vs = Vb + krow * HF_ + h * 16;
                        *(uint4*)&sK[pos * 16]     = *(const uint4*)ks;
                        *(uint4*)&sK[pos * 16 + 8] = *(const uint4*)(ks + 8);
                        *(uint4*)&sV[pos * 16]     = *(const uint4*)vs;
                        *(uint4*)&sV[pos * 16 + 8] = *(const uint4*)(vs + 8);
                    }
#endif
                    __syncthreads();

                    for (int dyi = 0; dyi < 5; ++dyi) {
                        const int pr = yc + dyi - 2 - py0;

                        // scores: Q x K_patch (2 WMMAs, key cols 0..19)
                        v8f s0 = {0.f,0.f,0.f,0.f,0.f,0.f,0.f,0.f};
                        v8f s1 = {0.f,0.f,0.f,0.f,0.f,0.f,0.f,0.f};
                        v16bf k0, k1;
                        const int n1c = lr + 16;
                        #pragma unroll
                        for (int i = 0; i < 8; ++i) {
                            k0[i] = sK[(pr * 20 + lr) * 16 + hi * 8 + i];
                            k1[i] = (n1c < 20) ? sK[(pr * 20 + n1c) * 16 + hi * 8 + i]
                                               : (__bf16)0.f;
                            k0[8 + i] = (__bf16)0.f;
                            k1[8 + i] = (__bf16)0.f;
                        }
                        s0 = wmma_bf16(qa, k0, s0);
                        s1 = wmma_bf16(qa, k1, s1);

                        // fixup + online softmax (C layout: q = r+8*hi, n = lane&15)
                        float alpha[8];
                        #pragma unroll
                        for (int r = 0; r < 8; ++r) {
                            const int M   = r + hi * 8;
                            const int xq  = x0 + M;
                            const int xcq = min(max(xq, 2), X_ - 3);
                            float sc0, sc1;
                            {
                                int dx = (px0 + lr) - xcq;
                                sc0 = (dx >= -2 && dx <= 2)
                                    ? (s0[r] + sQs[wave][M * 32 + dyi * 5 + dx + 2]) * scale
                                    : -1e30f;
                            }
                            {
                                int n  = lr + 16;
                                int dx = (px0 + n) - xcq;
                                sc1 = (n < 20 && dx >= -2 && dx <= 2)
                                    ? (s1[r] + sQs[wave][M * 32 + dyi * 5 + dx + 2]) * scale
                                    : -1e30f;
                            }
                            float cm = fmaxf(sc0, sc1);
                            #pragma unroll
                            for (int msk = 1; msk < 16; msk <<= 1)
                                cm = fmaxf(cm, __shfl_xor(cm, msk, 32));
                            const float mn = fmaxf(m8[r], cm);
                            const float a_ = __expf(m8[r] - mn);
                            const float e0 = __expf(sc0 - mn);
                            const float e1 = __expf(sc1 - mn);
                            float rs = e0 + e1;
                            #pragma unroll
                            for (int msk = 1; msk < 16; msk <<= 1)
                                rs += __shfl_xor(rs, msk, 32);
                            l8[r]    = l8[r] * a_ + rs;
                            m8[r]    = mn;
                            alpha[r] = a_;
                            sP[wave][M * 32 + lr]      = (__bf16)e0;
                            sP[wave][M * 32 + lr + 16] = (__bf16)e1;
                        }
                        #pragma unroll
                        for (int r = 0; r < 8; ++r) acc[r] *= alpha[r];

                        // P x V (1 WMMA): P via per-wave LDS transpose, V in B layout.
                        v16bf pa, vb;
                        #pragma unroll
                        for (int i = 0; i < 8; ++i) {
                            pa[i]     = sP[wave][lr * 32 + hi * 8 + i];
                            pa[8 + i] = sP[wave][lr * 32 + 16 + hi * 8 + i];
                            int n0 = hi * 8 + i;
                            int n1 = 16 + hi * 8 + i;
                            vb[i]     = sV[(pr * 20 + n0) * 16 + lr];
                            vb[8 + i] = (n1 < 20) ? sV[(pr * 20 + n1) * 16 + lr]
                                                  : (__bf16)0.f;
                        }
                        acc = wmma_bf16(pa, vb, acc);
                    } // dyi
                } // t

                #pragma unroll
                for (int r = 0; r < 8; ++r) {
                    const int M = r + hi * 8;
                    size_t orow = ((size_t)(tq * B_ + bb) * Y_ + y) * X_ + (x0 + M);
                    Ob[orow * HF_ + h * 16 + lr] = (__bf16)(acc[r] / l8[r]);
                }
            } // ri
        } // tq
    } // h
}

// ---------------------------------------------------------------------------
// Kernel 3: output projection out = attn(bf16) x Wout -> fp32, WMMA bf16.
// A-tile staged with a 1D TDM transfer (16KB contiguous bf16).
// ---------------------------------------------------------------------------
__global__ __launch_bounds__(256)
void out_proj_kernel(const __bf16* __restrict__ Ab,
                     const float*  __restrict__ Wout,
                     float* __restrict__ out)
{
    __shared__ __attribute__((aligned(16))) __bf16 sA[64 * 128];
    __shared__ __attribute__((aligned(16))) __bf16 sW[128 * 128];

    const int tid     = threadIdx.x;
    const int rowBase = blockIdx.x * 64;

#if USE_TDM
    if ((tid >> 5) == 0) {
        tdm_load_to_lds((unsigned)(size_t)(void*)&sA[0],
                        Ab + (size_t)rowBase * 128,
                        2048, 1, 1,  2048, 0, 0,
                        2048ull, 2048ull);
        __builtin_amdgcn_s_wait_tensorcnt(0);
    }
#else
    for (int idx = tid; idx < (64 * 128) / 8; idx += 256)
        ((uint4*)sA)[idx] = ((const uint4*)(Ab + (size_t)rowBase * 128))[idx];
#endif
    for (int idx = tid; idx < 128 * 128; idx += 256) {
        int n = idx >> 7, k = idx & 127;
        sW[idx] = (__bf16)Wout[k * 128 + n];
    }
    __syncthreads();

    const int wave = tid >> 5;
    const int lane = tid & 31;
    const int lr   = lane & 15;
    const int hi   = lane >> 4;

    for (int tile = wave; tile < 32; tile += 8) {
        int mt = tile >> 3, nt = tile & 7;
        v8f acc = {0.f, 0.f, 0.f, 0.f, 0.f, 0.f, 0.f, 0.f};
        #pragma unroll
        for (int kk = 0; kk < 4; ++kk) {
            const __bf16* ap = &sA[(mt * 16 + lr) * 128 + kk * 32 + hi * 8];
            const __bf16* bp = &sW[(nt * 16 + lr) * 128 + kk * 32 + hi * 8];
            v16bf a, b;
            #pragma unroll
            for (int i = 0; i < 8; ++i) {
                a[i] = ap[i];  a[8 + i] = ap[16 + i];
                b[i] = bp[i];  b[8 + i] = bp[16 + i];
            }
            acc = wmma_bf16(a, b, acc);
        }
        int col = nt * 16 + lr;
        #pragma unroll
        for (int r = 0; r < 8; ++r) {
            int row = rowBase + mt * 16 + r + hi * 8;
            out[(size_t)row * 128 + col] = acc[r];
        }
    }
}

// ---------------------------------------------------------------------------
// Host-side launch. Workspace (bf16): Q | K | V, 56.6MB each; attention
// output reuses the Q region in-place.
// ---------------------------------------------------------------------------
extern "C" void kernel_launch(void* const* d_in, const int* in_sizes, int n_in,
                              void* d_out, int out_size, void* d_ws, size_t ws_size,
                              hipStream_t stream)
{
    (void)in_sizes; (void)n_in; (void)out_size; (void)ws_size;

    const float* frames      = (const float*)d_in[0];
    const float* Wq          = (const float*)d_in[1];
    const float* Wk          = (const float*)d_in[2];
    const float* Wv          = (const float*)d_in[3];
    const float* Wout        = (const float*)d_in[4];
    const float* temp_emb    = (const float*)d_in[5];
    const float* spatial_emb = (const float*)d_in[6];

    __bf16* Qb = (__bf16*)d_ws;
    __bf16* Kb = Qb + (size_t)NROW * HF_;
    __bf16* Vb = Kb + (size_t)NROW * HF_;

    qkv_proj_kernel<<<NROW / 64, 256, 0, stream>>>(
        frames, Wq, Wk, Wv, temp_emb, spatial_emb, Qb, Kb, Vb);

    attn_kernel<<<B_ * 12 * 12, 256, 0, stream>>>(
        Qb, Kb, Vb, spatial_emb, /*Ob=*/Qb);

    out_proj_kernel<<<NROW / 64, 256, 0, stream>>>(
        Qb, Wout, (float*)d_out);
}
